// SerializedAttention_55825984913715
// MI455X (gfx1250) — compile-verified, matrix operand order checked by reading, not run on hardware
//
#include <hip/hip_runtime.h>
#include <hip/hip_bf16.h>

// ---------------- types ----------------
typedef __bf16 bf16_t;
typedef __attribute__((ext_vector_type(16))) __bf16 bf16x16;
typedef __attribute__((ext_vector_type(8)))  __bf16 bf16x8;
typedef __attribute__((ext_vector_type(4)))  __bf16 bf16x4;
typedef __attribute__((ext_vector_type(8)))  float  f32x8;

#define NPTS   65536
#define BATCH  4
#define CDIM   512
#define QKVW   1536     // 3*C
#define KPATCH 256
#define HEADS  8
#define DH     64
#define NCHUNK 256      // NPTS / KPATCH
#define SREAL  257      // K+1 (cls + patch)
#define SPAD   272      // 17*16, >= 257
#define KVPAD  288      // 9*32,  >= 257 (K-dim granularity 32 for PV)

// GEMM tiling: BM=64, BN=128, BK=64, 256 threads (8 waves).
#define BM 64
#define BN 128
#define BK 64

// ---------------- CDNA5 async global->LDS helpers ----------------
// GVS mode: mem = SADDR64 + VADDR32 (+offset), LDS = VDST32 (+offset).
// Tracked with ASYNCcnt; drain with s_wait_asynccnt before the barrier.
__device__ __forceinline__ unsigned lds_off32(const void* p) {
    // flat LDS addresses map to LDS via addr[31:0] truncation (ISA 10.2)
    return (unsigned)(uintptr_t)p;
}
__device__ __forceinline__ void async_copy_b128(unsigned lds_byte,
                                                const void* gbase,
                                                unsigned gbyte_off) {
    asm volatile("global_load_async_to_lds_b128 %0, %1, %2"
                 :: "v"(lds_byte), "v"(gbyte_off), "s"(gbase)
                 : "memory");
}
__device__ __forceinline__ void wait_async0() {
    asm volatile("s_wait_asynccnt 0x0" ::: "memory");
}

// ---------------- WMMA helpers ----------------
__device__ __forceinline__ f32x8 wmma_bf16(bf16x16 a, bf16x16 b, f32x8 c) {
    return __builtin_amdgcn_wmma_f32_16x16x32_bf16(
        /*neg_a=*/false, a, /*neg_b=*/false, b,
        /*c_mod=*/(short)0, c, /*reuse_a=*/false, /*reuse_b=*/false);
}

// Load a 16x32 bf16 fragment from a row-major [rows][stride] bf16 buffer.
// ISA 16-bit A layout: lane L holds row (L&15); per-lane elements cover
// K = base..base+7 and base+16..base+23 with base = (L<16 ? 0 : 8).
// Same layout serves as a B fragment when the buffer stores B transposed.
__device__ __forceinline__ bf16x16 load_frag(const bf16_t* base, int row0, int stride) {
    const int lane = threadIdx.x & 31;
    const int row  = row0 + (lane & 15);
    const int kh   = (lane & 16) ? 8 : 0;
    const bf16_t* p = base + row * stride + kh;
    bf16x8 lo = *(const bf16x8*)(p);        // K = kh .. kh+7
    bf16x8 hi = *(const bf16x8*)(p + 16);   // K = kh+16 .. kh+23
    return __builtin_shufflevector(lo, hi, 0,1,2,3,4,5,6,7,8,9,10,11,12,13,14,15);
}

// ------------- kernel 0: weight conversion fp32 -> bf16, TRANSPOSED -------
__global__ void convert_weights_kernel(const float* __restrict__ Wqkv,
                                       const float* __restrict__ Wproj,
                                       bf16_t* __restrict__ WqkvT,
                                       bf16_t* __restrict__ WprojT) {
    int i = blockIdx.x * 256 + threadIdx.x;
    if (i < QKVW * CDIM) {
        int n = i / CDIM, k = i % CDIM;
        WqkvT[i] = (bf16_t)Wqkv[(size_t)k * QKVW + n];
    }
    if (i < CDIM * CDIM) {
        int n = i >> 9, k = i & (CDIM - 1);
        WprojT[i] = (bf16_t)Wproj[(size_t)k * CDIM + n];
    }
}

// ---------------- kernel 1: cls_qkv = cls_tokens @ Wqkv + bqkv (fp32) -----
__global__ void cls_qkv_kernel(const float* __restrict__ cls,
                               const float* __restrict__ Wqkv,
                               const float* __restrict__ bqkv,
                               float* __restrict__ out) {
    int i = blockIdx.x * 256 + threadIdx.x;     // BATCH*QKVW = 6144
    if (i >= BATCH * QKVW) return;
    int b = i / QKVW, col = i % QKVW;
    float s = bqkv[col];
    for (int k = 0; k < CDIM; ++k)
        s = fmaf(cls[b * CDIM + k], Wqkv[(size_t)k * QKVW + col], s);
    out[i] = s;
}

// ---------------- kernel 2: qkv_s = feat[order] @ Wqkv + bqkv (bf16 out) --
__global__ void qkv_gemm_kernel(const float* __restrict__ feat,
                                const int* __restrict__ order,
                                const bf16_t* __restrict__ WT,    // [1536][512] transposed
                                const float* __restrict__ bias,   // [1536]
                                bf16_t* __restrict__ out) {       // [65536][1536]
    __shared__ alignas(16) bf16_t Alds[BM * BK];    // A tile  [m][k]  (8 KB)
    __shared__ alignas(16) bf16_t Btlds[BN * BK];   // B^T tile [n][k] (16 KB)
    __shared__ int rowidx[BM];

    const int tid = threadIdx.x;
    const int m0 = blockIdx.x * BM;
    const int n0 = blockIdx.y * BN;
    if (tid < BM) rowidx[tid] = order[m0 + tid];
    __syncthreads();

    const int w    = tid >> 5;
    const int lane = tid & 31;
    const int wm   = (w & 3) * 16;        // wave's M tile
    const int wn0  = (w >> 2) * 64;       // wave's 4-tile N strip base
    const unsigned bt_base = lds_off32(Btlds);
    f32x8 acc[4];
#pragma unroll
    for (int j = 0; j < 4; ++j) acc[j] = (f32x8){0.f,0.f,0.f,0.f,0.f,0.f,0.f,0.f};

    for (int k0 = 0; k0 < CDIM; k0 += BK) {
        __syncthreads();
        // stage B^T asynchronously (contiguous 16B chunks, no VGPR round-trip)
        for (int s = tid; s < BN * (BK / 8); s += 256) {
            int n = s >> 3, c8 = (s & 7) * 8;
            unsigned ldsb = bt_base + (unsigned)((n * BK + c8) * sizeof(bf16_t));
            unsigned goff = (unsigned)(((unsigned)(n0 + n) * CDIM + k0 + c8) * sizeof(bf16_t));
            async_copy_b128(ldsb, WT, goff);
        }
        // stage A: gathered rows, fp32 -> bf16 (64 rows x 64 cols)
        for (int s = tid; s < BM * (BK / 4); s += 256) {
            int r = s >> 4, c4 = (s & 15) * 4;
            const float* src = feat + (size_t)rowidx[r] * CDIM + k0 + c4;
            const float4 f = *(const float4*)src;
            if (k0 + BK < CDIM) __builtin_prefetch(src + BK, 0, 1);
            bf16_t* dst = Alds + r * BK + c4;
            dst[0] = (bf16_t)f.x; dst[1] = (bf16_t)f.y;
            dst[2] = (bf16_t)f.z; dst[3] = (bf16_t)f.w;
        }
        wait_async0();
        __syncthreads();
        bf16x16 a0 = load_frag(Alds,      wm, BK);
        bf16x16 a1 = load_frag(Alds + 32, wm, BK);
#pragma unroll
        for (int j = 0; j < 4; ++j) {
            bf16x16 b0 = load_frag(Btlds,      wn0 + j * 16, BK);
            bf16x16 b1 = load_frag(Btlds + 32, wn0 + j * 16, BK);
            acc[j] = wmma_bf16(a0, b0, acc[j]);
            acc[j] = wmma_bf16(a1, b1, acc[j]);
        }
    }

    // epilogue: C layout -> VGPR r: (row = r or 8+r, col = lane&15)
    const int colb  = lane & 15;
    const int rbase = (lane & 16) ? 8 : 0;
#pragma unroll
    for (int j = 0; j < 4; ++j) {
#pragma unroll
        for (int r = 0; r < 8; ++r) {
            int row = m0 + wm + rbase + r;
            int c   = n0 + wn0 + j * 16 + colb;
            out[(size_t)row * QKVW + c] = (bf16_t)(acc[j][r] + bias[c]);
        }
    }
}

// ---------------- kernel 3: per-(chunk, head) attention -------------------
__global__ void attn_kernel(const bf16_t* __restrict__ qkv,     // [65536][1536] serialized
                            const float*  __restrict__ clsqkv,  // [4][1536]
                            const int*    __restrict__ offset,  // [4]
                            bf16_t* __restrict__ attn_out,      // [65536][512] serialized
                            float*  __restrict__ cls_out) {     // [256][512]
    extern __shared__ __align__(16) char smem_raw[];
    bf16_t* Qs = (bf16_t*)smem_raw;        // [272][64]
    bf16_t* Ks = Qs + SPAD * DH;           // [272][64]
    bf16_t* Vt = Ks + SPAD * DH;           // [64][288] (transposed V)
    bf16_t* Ps = Vt + DH * KVPAD;          // 8 * [16][288] per-wave P buffers

    const int tid   = threadIdx.x;
    const int chunk = blockIdx.x >> 3;
    const int h     = blockIdx.x & 7;

    // batch id for this chunk (searchsorted(offset, chunk*K, right))
    const int start = chunk * KPATCH;
    int bt = 0;
#pragma unroll
    for (int j = 0; j < BATCH; ++j) bt += (offset[j] <= start) ? 1 : 0;

    // ---- stage Q and K: row 0 = cls (fp32 src), rows 1..256 = qkv, pad 0 ----
#pragma unroll
    for (int which = 0; which < 2; ++which) {
        const int qoff = which * CDIM;               // 0 for Q, 512 for K
        bf16_t* dst = which ? Ks : Qs;
        for (int idx = tid; idx < SPAD * (DH / 4); idx += 256) {
            int row = idx >> 4, c4 = (idx & 15) * 4;
            bf16_t v0, v1, v2, v3;
            if (row == 0) {
                const float4 f = *(const float4*)(clsqkv + bt * QKVW + qoff + h * DH + c4);
                v0 = (bf16_t)f.x; v1 = (bf16_t)f.y; v2 = (bf16_t)f.z; v3 = (bf16_t)f.w;
            } else if (row <= KPATCH) {
                const bf16x4 f = *(const bf16x4*)(qkv + (size_t)(chunk * KPATCH + row - 1) * QKVW
                                                  + qoff + h * DH + c4);
                v0 = f[0]; v1 = f[1]; v2 = f[2]; v3 = f[3];
            } else {
                v0 = v1 = v2 = v3 = (bf16_t)0.f;
            }
            bf16_t* d = dst + row * DH + c4;
            d[0] = v0; d[1] = v1; d[2] = v2; d[3] = v3;
        }
    }
    // ---- stage V transposed: coalesced row-major reads, LDS-side scatter ----
    for (int idx = tid; idx < SPAD * (DH / 4); idx += 256) {
        int key = idx >> 4, c4 = (idx & 15) * 4;
        bf16_t v0, v1, v2, v3;
        if (key == 0) {
            const float4 f = *(const float4*)(clsqkv + bt * QKVW + 2 * CDIM + h * DH + c4);
            v0 = (bf16_t)f.x; v1 = (bf16_t)f.y; v2 = (bf16_t)f.z; v3 = (bf16_t)f.w;
        } else if (key <= KPATCH) {
            const bf16x4 f = *(const bf16x4*)(qkv + (size_t)(chunk * KPATCH + key - 1) * QKVW
                                              + 2 * CDIM + h * DH + c4);
            v0 = f[0]; v1 = f[1]; v2 = f[2]; v3 = f[3];
        } else {
            v0 = v1 = v2 = v3 = (bf16_t)0.f;
        }
        Vt[(c4 + 0) * KVPAD + key] = v0;
        Vt[(c4 + 1) * KVPAD + key] = v1;
        Vt[(c4 + 2) * KVPAD + key] = v2;
        Vt[(c4 + 3) * KVPAD + key] = v3;
    }
    // zero Vt keys 272..287
    for (int idx = tid; idx < DH * (KVPAD - SPAD); idx += 256) {
        int d = idx >> 4, c = idx & 15;
        Vt[d * KVPAD + SPAD + c] = (bf16_t)0.f;
    }
    __syncthreads();

    const int w     = tid >> 5;
    const int lane  = tid & 31;
    const int col_l = lane & 15;
    const int rbase = (lane & 16) ? 8 : 0;
    bf16_t* Pw = Ps + w * 16 * KVPAD;
    const float NEG_INF = -__builtin_inff();

    for (int t = w; t < SPAD / 16; t += 8) {      // 17 query tiles over 8 waves
        // ---- S = Q K^T : 16 x 272, fp32 ----
        bf16x16 qa0 = load_frag(Qs,      t * 16, DH);
        bf16x16 qa1 = load_frag(Qs + 32, t * 16, DH);
        f32x8 S[17];
#pragma unroll
        for (int n = 0; n < 17; ++n) {
            f32x8 s = {0.f,0.f,0.f,0.f,0.f,0.f,0.f,0.f};
            bf16x16 kb0 = load_frag(Ks,      n * 16, DH);
            bf16x16 kb1 = load_frag(Ks + 32, n * 16, DH);
            s = wmma_bf16(qa0, kb0, s);
            s = wmma_bf16(qa1, kb1, s);
            S[n] = s;
        }
        // ---- scale + key mask ----
#pragma unroll
        for (int n = 0; n < 17; ++n) {
            int col = n * 16 + col_l;
#pragma unroll
            for (int r = 0; r < 8; ++r)
                S[n][r] = (col < SREAL) ? S[n][r] * 0.125f : NEG_INF;
        }
        // ---- row max (per VGPR slot r, reduced across 16-lane half) ----
        float mx[8], sm[8];
#pragma unroll
        for (int r = 0; r < 8; ++r) {
            float m = NEG_INF;
#pragma unroll
            for (int n = 0; n < 17; ++n) m = fmaxf(m, S[n][r]);
            m = fmaxf(m, __shfl_xor(m, 1, 32));
            m = fmaxf(m, __shfl_xor(m, 2, 32));
            m = fmaxf(m, __shfl_xor(m, 4, 32));
            m = fmaxf(m, __shfl_xor(m, 8, 32));
            mx[r] = m;
        }
        // ---- exp + row sum ----
#pragma unroll
        for (int r = 0; r < 8; ++r) {
            float s = 0.f;
#pragma unroll
            for (int n = 0; n < 17; ++n) {
                float e = __expf(S[n][r] - mx[r]);
                S[n][r] = e;
                s += e;
            }
            s += __shfl_xor(s, 1, 32);
            s += __shfl_xor(s, 2, 32);
            s += __shfl_xor(s, 4, 32);
            s += __shfl_xor(s, 8, 32);
            sm[r] = s;
        }
        float inv[8];
#pragma unroll
        for (int r = 0; r < 8; ++r) inv[r] = 1.f / sm[r];

        // ---- write normalized P (bf16) into this wave's LDS buffer ----
#pragma unroll
        for (int n = 0; n < 17; ++n) {
#pragma unroll
            for (int r = 0; r < 8; ++r)
                Pw[(rbase + r) * KVPAD + n * 16 + col_l] = (bf16_t)(S[n][r] * inv[r]);
        }
        // zero-pad P cols 272..287
        {
            int zr = lane & 15;
            int zc = SPAD + ((lane & 16) ? 8 : 0);
#pragma unroll
            for (int j = 0; j < 8; ++j) Pw[zr * KVPAD + zc + j] = (bf16_t)0.f;
        }

        // ---- O = P @ V : 16 x 64, fp32 ----
        f32x8 O[4];
#pragma unroll
        for (int n = 0; n < 4; ++n) O[n] = (f32x8){0.f,0.f,0.f,0.f,0.f,0.f,0.f,0.f};
#pragma unroll
        for (int s = 0; s < KVPAD / 32; ++s) {       // 9 K-steps of 32 keys
            bf16x16 pa = load_frag(Pw + s * 32, 0, KVPAD);
#pragma unroll
            for (int n = 0; n < 4; ++n) {
                bf16x16 vb = load_frag(Vt + s * 32, n * 16, KVPAD);
                O[n] = wmma_bf16(pa, vb, O[n]);
            }
        }
        // ---- scatter O: row 0 -> cls_out (fp32), rows 1..256 -> attn_out ----
#pragma unroll
        for (int n = 0; n < 4; ++n) {
            int c = h * DH + n * 16 + col_l;
#pragma unroll
            for (int r = 0; r < 8; ++r) {
                int seqrow = t * 16 + rbase + r;
                float val = O[n][r];
                if (seqrow == 0) {
                    cls_out[chunk * CDIM + c] = val;
                } else if (seqrow <= KPATCH) {
                    attn_out[(size_t)(chunk * KPATCH + seqrow - 1) * CDIM + c] = (bf16_t)val;
                }
            }
        }
    }
}

// ---------------- kernel 4: feat_out = attn_out[inverse] @ Wproj + bproj --
__global__ void proj_gemm_kernel(const bf16_t* __restrict__ X,      // [65536][512]
                                 const int* __restrict__ inverse,
                                 const bf16_t* __restrict__ WT,     // [512][512] transposed
                                 const float* __restrict__ bias,    // [512]
                                 float* __restrict__ out) {         // [65536][512]
    __shared__ alignas(16) bf16_t Alds[BM * BK];
    __shared__ alignas(16) bf16_t Btlds[BN * BK];
    __shared__ int rowidx[BM];

    const int tid = threadIdx.x;
    const int m0 = blockIdx.x * BM;
    const int n0 = blockIdx.y * BN;
    if (tid < BM) rowidx[tid] = inverse[m0 + tid];
    __syncthreads();

    const int w    = tid >> 5;
    const int lane = tid & 31;
    const int wm   = (w & 3) * 16;
    const int wn0  = (w >> 2) * 64;
    const unsigned a_base  = lds_off32(Alds);
    const unsigned bt_base = lds_off32(Btlds);
    f32x8 acc[4];
#pragma unroll
    for (int j = 0; j < 4; ++j) acc[j] = (f32x8){0.f,0.f,0.f,0.f,0.f,0.f,0.f,0.f};

    for (int k0 = 0; k0 < CDIM; k0 += BK) {
        __syncthreads();
        // stage A asynchronously: gathered bf16 rows (64 x 64), 16B chunks
        for (int s = tid; s < BM * (BK / 8); s += 256) {
            int r = s >> 3, c8 = (s & 7) * 8;
            unsigned ldsb = a_base + (unsigned)((r * BK + c8) * sizeof(bf16_t));
            unsigned goff = (unsigned)(((unsigned)rowidx[r] * CDIM + k0 + c8) * sizeof(bf16_t));
            async_copy_b128(ldsb, X, goff);
        }
        // stage B^T asynchronously from pre-transposed weights
        for (int s = tid; s < BN * (BK / 8); s += 256) {
            int n = s >> 3, c8 = (s & 7) * 8;
            unsigned ldsb = bt_base + (unsigned)((n * BK + c8) * sizeof(bf16_t));
            unsigned goff = (unsigned)(((unsigned)(n0 + n) * CDIM + k0 + c8) * sizeof(bf16_t));
            async_copy_b128(ldsb, WT, goff);
        }
        wait_async0();
        __syncthreads();
        bf16x16 a0 = load_frag(Alds,      wm, BK);
        bf16x16 a1 = load_frag(Alds + 32, wm, BK);
#pragma unroll
        for (int j = 0; j < 4; ++j) {
            bf16x16 b0 = load_frag(Btlds,      wn0 + j * 16, BK);
            bf16x16 b1 = load_frag(Btlds + 32, wn0 + j * 16, BK);
            acc[j] = wmma_bf16(a0, b0, acc[j]);
            acc[j] = wmma_bf16(a1, b1, acc[j]);
        }
    }

    const int colb  = lane & 15;
    const int rbase = (lane & 16) ? 8 : 0;
#pragma unroll
    for (int j = 0; j < 4; ++j) {
#pragma unroll
        for (int r = 0; r < 8; ++r) {
            int row = m0 + wm + rbase + r;
            int c   = n0 + wn0 + j * 16 + colb;
            out[(size_t)row * CDIM + c] = acc[j][r] + bias[c];
        }
    }
}

// ---------------- kernel 5: cls_feat = per-batch mean of cls_out ----------
__global__ void cls_agg_kernel(const float* __restrict__ cls_out,  // [256][512]
                               const int* __restrict__ offset,
                               float* __restrict__ out) {          // [4][512]
    int i = blockIdx.x * 256 + threadIdx.x;      // BATCH*CDIM = 2048
    if (i >= BATCH * CDIM) return;
    int b = i >> 9, col = i & (CDIM - 1);
    float s = 0.f;
    int cnt = 0;
    for (int ch = 0; ch < NCHUNK; ++ch) {
        int st = ch * KPATCH;
        int bt = 0;
#pragma unroll
        for (int j = 0; j < BATCH; ++j) bt += (offset[j] <= st) ? 1 : 0;
        if (bt == b) { s += cls_out[ch * CDIM + col]; ++cnt; }
    }
    out[i] = (cnt > 0) ? s / (float)cnt : 0.f;
}

// ---------------- launch ----------------
extern "C" void kernel_launch(void* const* d_in, const int* in_sizes, int n_in,
                              void* d_out, int out_size, void* d_ws, size_t ws_size,
                              hipStream_t stream) {
    const float* feat   = (const float*)d_in[0];
    const float* cls    = (const float*)d_in[1];
    const float* Wqkv   = (const float*)d_in[2];
    const float* bqkv   = (const float*)d_in[3];
    const float* Wproj  = (const float*)d_in[4];
    const float* bproj  = (const float*)d_in[5];
    const int*   order  = (const int*)d_in[6];
    const int*   inverse= (const int*)d_in[7];
    const int*   offset = (const int*)d_in[8];
    float* out = (float*)d_out;

    // workspace carve-out (256B aligned regions)
    char* ws = (char*)d_ws;
    size_t off = 0;
    auto carve = [&](size_t bytes) -> void* {
        void* p = ws + off;
        off += (bytes + 255) & ~(size_t)255;
        return p;
    };
    bf16_t* WqkvT   = (bf16_t*)carve((size_t)QKVW * CDIM * 2);
    bf16_t* WprojT  = (bf16_t*)carve((size_t)CDIM * CDIM * 2);
    float*  clsqkv  = (float*) carve((size_t)BATCH * QKVW * 4);
    bf16_t* qkv_s   = (bf16_t*)carve((size_t)NPTS * QKVW * 2);
    bf16_t* attnout = (bf16_t*)carve((size_t)NPTS * CDIM * 2);
    float*  clsout  = (float*) carve((size_t)NCHUNK * CDIM * 4);

    const size_t ATTN_SMEM =
        (size_t)(SPAD * DH * 2 + DH * KVPAD + 8 * 16 * KVPAD) * sizeof(bf16_t);
    (void)hipFuncSetAttribute((const void*)attn_kernel,
                              hipFuncAttributeMaxDynamicSharedMemorySize,
                              (int)ATTN_SMEM);

    convert_weights_kernel<<<(QKVW * CDIM + 255) / 256, 256, 0, stream>>>(
        Wqkv, Wproj, WqkvT, WprojT);
    cls_qkv_kernel<<<(BATCH * QKVW + 255) / 256, 256, 0, stream>>>(
        cls, Wqkv, bqkv, clsqkv);
    qkv_gemm_kernel<<<dim3(NPTS / BM, QKVW / BN), 256, 0, stream>>>(
        feat, order, WqkvT, bqkv, qkv_s);
    attn_kernel<<<NCHUNK * HEADS, 256, ATTN_SMEM, stream>>>(
        qkv_s, clsqkv, offset, attnout, clsout);
    proj_gemm_kernel<<<dim3(NPTS / BM, CDIM / BN), 256, 0, stream>>>(
        attnout, inverse, WprojT, bproj, out);
    cls_agg_kernel<<<(BATCH * CDIM + 255) / 256, 256, 0, stream>>>(
        clsout, offset, out + (size_t)NPTS * CDIM);
}
